// GptOssDecoderLayer_19980187861077
// MI455X (gfx1250) — compile-verified, matrix operand order checked by reading, not run on hardware
//
#include <hip/hip_runtime.h>
#include <hip/hip_bf16.h>
#include <cmath>

// ---------------- problem constants ----------------
#define BB 2
#define SS 1024
#define HH 1024
#define NHEADS 16
#define HD 64
#define KVHEADS 4
#define GROUPS (NHEADS / KVHEADS)         // 4
#define II 2048
#define EE 16
#define TOPK 4
#define TT (BB * SS)                      // 2048 tokens
#define QKVW (NHEADS*HD + 2*KVHEADS*HD)   // 1536
#define KOFF (NHEADS*HD)                  // 1024
#define VOFF (NHEADS*HD + KVHEADS*HD)     // 1280
#define MAXSLOTS (TT * TOPK)              // 8192
#define MAXPAD (MAXSLOTS + EE*64)         // 9216 (64-aligned per-expert regions)
#define MAXROWT (MAXPAD / 64)             // 144 64-row tiles
#define NEGV (-1.0e9f)
#define EPS 1e-5f

typedef __attribute__((ext_vector_type(16))) _Float16 v16h;
typedef __attribute__((ext_vector_type(8)))  float    v8f;

#define WMMA_F16(a, b, c) \
  __builtin_amdgcn_wmma_f32_16x16x32_f16(false, (a), false, (b), (short)0, (c), false, false)

// ---------------- WMMA fragment loaders ----------------
// A fragment (16x32 f16, row-major source, stride lda):
//   lane L: row = L&15; K-halves: lanes<16 hold K 0..7 & 16..23, lanes>=16 hold K 8..15 & 24..31
__device__ __forceinline__ v16h frag_A_f16(const _Float16* __restrict__ A, int lda) {
  int lane = threadIdx.x & 31;
  int row = lane & 15;
  int kb  = (lane & 16) ? 8 : 0;
  const _Float16* p = A + (size_t)row * lda + kb;
  v16h a;
#pragma unroll
  for (int i = 0; i < 8; ++i) a[i] = p[i];
#pragma unroll
  for (int i = 0; i < 8; ++i) a[8 + i] = p[16 + i];
  return a;
}

__device__ __forceinline__ v16h frag_A_f32(const float* __restrict__ A, int lda) {
  int lane = threadIdx.x & 31;
  int row = lane & 15;
  int kb  = (lane & 16) ? 8 : 0;
  const float* p = A + (size_t)row * lda + kb;
  v16h a;
#pragma unroll
  for (int i = 0; i < 8; ++i) a[i] = (_Float16)p[i];
#pragma unroll
  for (int i = 0; i < 8; ++i) a[8 + i] = (_Float16)p[16 + i];
  return a;
}

// gathered A row: each lane supplies its own row's token (tok<0 -> zero row)
__device__ __forceinline__ v16h frag_A_gather_f16(const _Float16* __restrict__ base,
                                                  int tok, int lda, int kk) {
  int lane = threadIdx.x & 31;
  int kb = (lane & 16) ? 8 : 0;
  v16h a = {};
  if (tok >= 0) {
    const _Float16* p = base + (size_t)tok * lda + kk + kb;
#pragma unroll
    for (int i = 0; i < 8; ++i) a[i] = p[i];
#pragma unroll
    for (int i = 0; i < 8; ++i) a[8 + i] = p[16 + i];
  }
  return a;
}

// B fragment (32x16, source is K x N row-major f32, stride ldb):
//   lane L: col = L&15; same K-half pattern as A
__device__ __forceinline__ v16h frag_B_f32(const float* __restrict__ Bm, int ldb) {
  int lane = threadIdx.x & 31;
  int col = lane & 15;
  int kb  = (lane & 16) ? 8 : 0;
  v16h b;
#pragma unroll
  for (int i = 0; i < 8; ++i) b[i] = (_Float16)Bm[(size_t)(kb + i) * ldb + col];
#pragma unroll
  for (int i = 0; i < 8; ++i) b[8 + i] = (_Float16)Bm[(size_t)(16 + kb + i) * ldb + col];
  return b;
}

// B fragment when B is stored transposed (N x K row-major), i.e. B[k][n] = Bt[n][k]
__device__ __forceinline__ v16h frag_BT_f32(const float* __restrict__ Bt, int ldb) {
  int lane = threadIdx.x & 31;
  int col = lane & 15;
  int kb  = (lane & 16) ? 8 : 0;
  const float* p = Bt + (size_t)col * ldb + kb;
  v16h b;
#pragma unroll
  for (int i = 0; i < 8; ++i) b[i] = (_Float16)p[i];
#pragma unroll
  for (int i = 0; i < 8; ++i) b[8 + i] = (_Float16)p[16 + i];
  return b;
}

// ---------------- kernels ----------------

__global__ void k_init(int* counts, int* tileExpert, int* permTok) {
  int tid = threadIdx.x;
  if (tid < EE) counts[tid] = 0;
  for (int i = tid; i < MAXROWT; i += 256) tileExpert[i] = -1;
  for (int i = tid; i < MAXPAD; i += 256) permTok[i] = -1;
}

// RMSNorm (ln1): x f32 -> h f16
__global__ void k_rms1(const float* __restrict__ x, const float* __restrict__ w,
                       _Float16* __restrict__ out) {
  __shared__ float red[256];
  int t = blockIdx.x, tid = threadIdx.x;
  const float* xp = x + (size_t)t * HH;
  float ss = 0.f;
  for (int i = tid; i < HH; i += 256) { float v = xp[i]; ss += v * v; }
  red[tid] = ss; __syncthreads();
  for (int s = 128; s > 0; s >>= 1) { if (tid < s) red[tid] += red[tid + s]; __syncthreads(); }
  float rs = rsqrtf(red[0] / (float)HH + EPS);
  for (int i = tid; i < HH; i += 256)
    out[(size_t)t * HH + i] = (_Float16)(xp[i] * rs * w[i]);
}

// QKV projection: 64 rows x 16 cols per wave; B fragment reused across 4 M-subtiles
__global__ __launch_bounds__(32) void k_qkv(const _Float16* __restrict__ h16,
    const float* __restrict__ qw, const float* __restrict__ qb,
    const float* __restrict__ kw, const float* __restrict__ kbias,
    const float* __restrict__ vw, const float* __restrict__ vb,
    float* __restrict__ qkv) {
  int m0 = blockIdx.x * 64;
  int n0 = blockIdx.y * 16;
  const float *W, *Bv; int ldw, wn;
  if (n0 < KOFF)      { W = qw; Bv = qb;    ldw = NHEADS * HD;  wn = n0; }
  else if (n0 < VOFF) { W = kw; Bv = kbias; ldw = KVHEADS * HD; wn = n0 - KOFF; }
  else                { W = vw; Bv = vb;    ldw = KVHEADS * HD; wn = n0 - VOFF; }
  int lane = threadIdx.x, col = lane & 15, rhi = (lane & 16) ? 8 : 0;
  const _Float16* A = h16 + (size_t)m0 * HH;
  v8f z = {};
  v8f c[4] = {z, z, z, z};
  for (int kk = 0; kk < HH; kk += 32) {
    if (kk + 32 < HH)
      __builtin_prefetch((const void*)(W + (size_t)(kk + 32) * ldw + wn), 0, 0);
    v16h b = frag_B_f32(W + (size_t)kk * ldw + wn, ldw);
#pragma unroll
    for (int s = 0; s < 4; ++s) {
      v16h a = frag_A_f16(A + (size_t)(s * 16) * HH + kk, HH);
      c[s] = WMMA_F16(a, b, c[s]);
    }
  }
  float bias = Bv[wn + col];
#pragma unroll
  for (int s = 0; s < 4; ++s)
#pragma unroll
    for (int r = 0; r < 8; ++r)
      qkv[(size_t)(m0 + s * 16 + r + rhi) * QKVW + n0 + col] = c[s][r] + bias;
}

// Flash attention: one wave per (16 q rows, head, batch); causal; GQA kv = h/GROUPS
__global__ __launch_bounds__(32) void k_attn(const float* __restrict__ qkv,
                                             _Float16* __restrict__ attn16) {
  __shared__ float    sc[16][32];
  __shared__ _Float16 p16[16][32];
  __shared__ float    scl[16];
  __shared__ float    linv[16];
  int lane = threadIdx.x;
  int q0 = blockIdx.x * 16;
  int h  = blockIdx.y;
  int b  = blockIdx.z;
  int kv = h / GROUPS;
  const float* qbase = qkv + (size_t)(b * SS + q0) * QKVW + h * HD;
  const float* kcol  = qkv + (size_t)(b * SS) * QKVW + KOFF + kv * HD;
  const float* vcol  = qkv + (size_t)(b * SS) * QKVW + VOFF + kv * HD;
  int col = lane & 15, rhi = (lane & 16) ? 8 : 0;

  v8f o0 = {}, o1 = {}, o2 = {}, o3 = {};
  float mrow = -INFINITY, lrow = 0.f;   // per-row state owned by lanes 0..15
  v16h qa0 = frag_A_f32(qbase, QKVW);        // K features 0..31
  v16h qa1 = frag_A_f32(qbase + 32, QKVW);   // K features 32..63
  int ktmax = (q0 + 15) >> 5;                // last 32-key tile touching the causal region
  for (int kt = 0; kt <= ktmax; ++kt) {
    int k0 = kt * 32;
    // scores = Q K^T for 32 keys (two 16x16 tiles)
    v8f s0 = {}, s1 = {};
    {
      const float* kb0 = kcol + (size_t)k0 * QKVW;
      s0 = WMMA_F16(qa0, frag_BT_f32(kb0, QKVW), s0);
      s0 = WMMA_F16(qa1, frag_BT_f32(kb0 + 32, QKVW), s0);
      const float* kb1 = kcol + (size_t)(k0 + 16) * QKVW;
      s1 = WMMA_F16(qa0, frag_BT_f32(kb1, QKVW), s1);
      s1 = WMMA_F16(qa1, frag_BT_f32(kb1 + 32, QKVW), s1);
    }
    // scale + causal mask -> LDS
#pragma unroll
    for (int r = 0; r < 8; ++r) {
      int m = r + rhi;
      int qi = q0 + m;
      sc[m][col]      = ((k0 + col)      <= qi) ? s0[r] * 0.125f : NEGV;
      sc[m][16 + col] = ((k0 + 16 + col) <= qi) ? s1[r] * 0.125f : NEGV;
    }
    __syncthreads();
    // online softmax, one row per lane (lanes 0..15)
    if (lane < 16) {
      int row = lane;
      float mx = mrow;
#pragma unroll
      for (int j = 0; j < 32; ++j) mx = fmaxf(mx, sc[row][j]);
      float sum = 0.f;
#pragma unroll
      for (int j = 0; j < 32; ++j) {
        float p = __expf(sc[row][j] - mx);
        p16[row][j] = (_Float16)p;
        sum += p;
      }
      float so = __expf(mrow - mx);
      lrow = lrow * so + sum;
      mrow = mx;
      scl[row] = so;
    }
    __syncthreads();
    // rescale O accumulators
    float sv[8];
#pragma unroll
    for (int r = 0; r < 8; ++r) sv[r] = scl[r + rhi];
#pragma unroll
    for (int r = 0; r < 8; ++r) { o0[r] *= sv[r]; o1[r] *= sv[r]; o2[r] *= sv[r]; o3[r] *= sv[r]; }
    // O += P @ V (K-dim = 32 keys)
    v16h pa = frag_A_f16(&p16[0][0], 32);
    const float* vb = vcol + (size_t)k0 * QKVW;
    o0 = WMMA_F16(pa, frag_B_f32(vb +  0, QKVW), o0);
    o1 = WMMA_F16(pa, frag_B_f32(vb + 16, QKVW), o1);
    o2 = WMMA_F16(pa, frag_B_f32(vb + 32, QKVW), o2);
    o3 = WMMA_F16(pa, frag_B_f32(vb + 48, QKVW), o3);
    __syncthreads();
  }
  if (lane < 16) linv[lane] = 1.f / lrow;
  __syncthreads();
  float iv[8];
#pragma unroll
  for (int r = 0; r < 8; ++r) iv[r] = linv[r + rhi];
  _Float16* ob = attn16 + (size_t)(b * SS + q0) * (NHEADS * HD) + h * HD;
#pragma unroll
  for (int r = 0; r < 8; ++r) {
    size_t rowoff = (size_t)(r + rhi) * (NHEADS * HD);
    ob[rowoff +  0 + col] = (_Float16)(o0[r] * iv[r]);
    ob[rowoff + 16 + col] = (_Float16)(o1[r] * iv[r]);
    ob[rowoff + 32 + col] = (_Float16)(o2[r] * iv[r]);
    ob[rowoff + 48 + col] = (_Float16)(o3[r] * iv[r]);
  }
}

// O projection + residual: 64 rows x 16 cols per wave
__global__ __launch_bounds__(32) void k_oproj(const _Float16* __restrict__ attn16,
    const float* __restrict__ ow, const float* __restrict__ obias,
    const float* __restrict__ resid, float* __restrict__ x2) {
  int m0 = blockIdx.x * 64, n0 = blockIdx.y * 16;
  int lane = threadIdx.x, col = lane & 15, rhi = (lane & 16) ? 8 : 0;
  const _Float16* A = attn16 + (size_t)m0 * (NHEADS * HD);
  v8f z = {};
  v8f c[4] = {z, z, z, z};
  for (int kk = 0; kk < NHEADS * HD; kk += 32) {
    if (kk + 32 < NHEADS * HD)
      __builtin_prefetch((const void*)(ow + (size_t)(kk + 32) * HH + n0), 0, 0);
    v16h b = frag_B_f32(ow + (size_t)kk * HH + n0, HH);
#pragma unroll
    for (int s = 0; s < 4; ++s) {
      v16h a = frag_A_f16(A + (size_t)(s * 16) * (NHEADS * HD) + kk, NHEADS * HD);
      c[s] = WMMA_F16(a, b, c[s]);
    }
  }
  float bias = obias[n0 + col];
#pragma unroll
  for (int s = 0; s < 4; ++s)
#pragma unroll
    for (int r = 0; r < 8; ++r) {
      size_t idx = (size_t)(m0 + s * 16 + r + rhi) * HH + n0 + col;
      x2[idx] = resid[idx] + c[s][r] + bias;
    }
}

// RMSNorm(ln2) + router logits + top-4 softmax + expert histogram
__global__ void k_router(const float* __restrict__ x2, const float* __restrict__ w2,
                         const float* __restrict__ rw, const float* __restrict__ rb,
                         _Float16* __restrict__ h2, int* __restrict__ topidx,
                         float* __restrict__ topw, int* __restrict__ counts) {
  __shared__ float red[256];
  __shared__ float logits[EE];
  int t = blockIdx.x, tid = threadIdx.x;
  const float* x = x2 + (size_t)t * HH;
  float ss = 0.f;
  for (int i = tid; i < HH; i += 256) { float v = x[i]; ss += v * v; }
  red[tid] = ss; __syncthreads();
  for (int s = 128; s > 0; s >>= 1) { if (tid < s) red[tid] += red[tid + s]; __syncthreads(); }
  float rs = rsqrtf(red[0] / (float)HH + EPS);
  __syncthreads();
  float le[EE];
#pragma unroll
  for (int e = 0; e < EE; ++e) le[e] = 0.f;
  for (int i = tid; i < HH; i += 256) {
    float hv = x[i] * rs * w2[i];
    h2[(size_t)t * HH + i] = (_Float16)hv;
    const float* rp = rw + (size_t)i * EE;
#pragma unroll
    for (int e = 0; e < EE; ++e) le[e] += hv * rp[e];
  }
  __syncthreads();
  for (int e = 0; e < EE; ++e) {
    red[tid] = le[e]; __syncthreads();
    for (int s = 128; s > 0; s >>= 1) { if (tid < s) red[tid] += red[tid + s]; __syncthreads(); }
    if (tid == 0) logits[e] = red[0] + rb[e];
    __syncthreads();
  }
  if (tid == 0) {
    float v[EE];
#pragma unroll
    for (int e = 0; e < EE; ++e) v[e] = logits[e];
    int idx[TOPK]; float tv[TOPK];
    for (int s = 0; s < TOPK; ++s) {
      int best = 0; float bv = v[0];
      for (int e = 1; e < EE; ++e) if (v[e] > bv) { bv = v[e]; best = e; }
      idx[s] = best; tv[s] = bv; v[best] = -INFINITY;
    }
    float ex[TOPK], denom = 0.f;
    for (int s = 0; s < TOPK; ++s) { ex[s] = __expf(tv[s] - tv[0]); denom += ex[s]; }
    for (int s = 0; s < TOPK; ++s) {
      topidx[t * TOPK + s] = idx[s];
      topw[t * TOPK + s] = ex[s] / denom;
      atomicAdd(&counts[idx[s]], 1);
    }
  }
}

// prefix-sum (64-aligned) over expert counts; mark 64-row-tile -> expert map
__global__ void k_offsets(const int* __restrict__ counts, int* __restrict__ cursor,
                          int* __restrict__ tileExpert) {
  if (threadIdx.x == 0 && blockIdx.x == 0) {
    int off = 0;
    for (int e = 0; e < EE; ++e) {
      cursor[e] = off;
      int padded = (counts[e] + 63) & ~63;
      int t0 = off >> 6;
      for (int j = 0; j < (padded >> 6); ++j) tileExpert[t0 + j] = e;
      off += padded;
    }
  }
}

// scatter (token,slot) into its expert's row region
__global__ void k_scatter(const int* __restrict__ topidx, int* __restrict__ cursor,
                          int* __restrict__ permTok, int* __restrict__ slotPos) {
  int idx = blockIdx.x * 256 + threadIdx.x;   // token*TOPK + slot
  int e = topidx[idx];
  int pos = atomicAdd(&cursor[e], 1);
  permTok[pos] = idx;
  slotPos[idx] = pos;
}

// MoE GEMM1: 64 gathered rows x (gate,up) 16-col pair per wave; B fragments reused 4x
__global__ __launch_bounds__(32) void k_moe1(const _Float16* __restrict__ h2,
    const int* __restrict__ permTok, const int* __restrict__ tileExpert,
    const float* __restrict__ gupw, const float* __restrict__ gupb,
    _Float16* __restrict__ inter16) {
  int rt = blockIdx.x;                // 64-row tile
  int e = tileExpert[rt];
  if (e < 0) return;
  int n0 = blockIdx.y * 16;
  int lane = threadIdx.x, col = lane & 15, rhi = (lane & 16) ? 8 : 0;
  int tok[4];
#pragma unroll
  for (int s = 0; s < 4; ++s) {
    int pr = permTok[rt * 64 + s * 16 + (lane & 15)];
    tok[s] = (pr >= 0) ? (pr >> 2) : -1;   // TOPK == 4
  }
  const float* wg = gupw + (size_t)e * HH * (2 * II) + n0;
  v8f z = {};
  v8f cg[4] = {z, z, z, z};
  v8f cu[4] = {z, z, z, z};
  for (int kk = 0; kk < HH; kk += 32) {
    const float* wgk = wg + (size_t)kk * (2 * II);
    if (kk + 32 < HH) __builtin_prefetch((const void*)(wgk + (size_t)32 * (2 * II)), 0, 0);
    v16h bg = frag_B_f32(wgk, 2 * II);
    v16h bu = frag_B_f32(wgk + II, 2 * II);
#pragma unroll
    for (int s = 0; s < 4; ++s) {
      v16h a = frag_A_gather_f16(h2, tok[s], HH, kk);
      cg[s] = WMMA_F16(a, bg, cg[s]);
      cu[s] = WMMA_F16(a, bu, cu[s]);
    }
  }
  float bg = gupb[(size_t)e * 2 * II + n0 + col];
  float bu = gupb[(size_t)e * 2 * II + II + n0 + col];
  _Float16* ob = inter16 + (size_t)(rt * 64) * II + n0;
#pragma unroll
  for (int s = 0; s < 4; ++s)
#pragma unroll
    for (int r = 0; r < 8; ++r) {
      float g = cg[s][r] + bg, u = cu[s][r] + bu;
      float inter = (g * (1.f / (1.f + __expf(-g)))) * u;
      ob[(size_t)(s * 16 + r + rhi) * II + col] = (_Float16)inter;
    }
}

// MoE GEMM2: 64 rows x 16 cols per wave; slotOut = (inter @ down_w + down_b) * combine_weight
__global__ __launch_bounds__(32) void k_moe2(const _Float16* __restrict__ inter16,
    const int* __restrict__ permTok, const int* __restrict__ tileExpert,
    const float* __restrict__ topw, const float* __restrict__ dw,
    const float* __restrict__ db, float* __restrict__ slotOut) {
  __shared__ float pw[64];
  __shared__ int   pv[64];
  int rt = blockIdx.x;
  int e = tileExpert[rt];
  if (e < 0) return;
  int n0 = blockIdx.y * 16;
  int lane = threadIdx.x, col = lane & 15, rhi = (lane & 16) ? 8 : 0;
  for (int i = lane; i < 64; i += 32) {
    int v = permTok[rt * 64 + i];
    pv[i] = v;
    pw[i] = (v >= 0) ? topw[v] : 0.f;
  }
  __syncthreads();
  const _Float16* A = inter16 + (size_t)(rt * 64) * II;
  const float* W = dw + (size_t)e * II * HH + n0;
  v8f z = {};
  v8f c[4] = {z, z, z, z};
  for (int kk = 0; kk < II; kk += 32) {
    if (kk + 32 < II) __builtin_prefetch((const void*)(W + (size_t)(kk + 32) * HH), 0, 0);
    v16h b = frag_B_f32(W + (size_t)kk * HH, HH);
#pragma unroll
    for (int s = 0; s < 4; ++s) {
      v16h a = frag_A_f16(A + (size_t)(s * 16) * II + kk, II);
      c[s] = WMMA_F16(a, b, c[s]);
    }
  }
  float bias = db[(size_t)e * HH + n0 + col];
  float* ob = slotOut + (size_t)(rt * 64) * HH + n0;
#pragma unroll
  for (int s = 0; s < 4; ++s)
#pragma unroll
    for (int r = 0; r < 8; ++r) {
      int m = s * 16 + r + rhi;
      if (pv[m] >= 0) ob[(size_t)m * HH + col] = (c[s][r] + bias) * pw[m];
    }
}

// final: out = x2 + sum over the token's 4 expert-slot outputs (fixed order -> deterministic)
__global__ void k_combine(const float* __restrict__ x2, const float* __restrict__ slotOut,
                          const int* __restrict__ slotPos, float* __restrict__ out) {
  int t = blockIdx.x, tid = threadIdx.x;
  int p0 = slotPos[t * TOPK + 0];
  int p1 = slotPos[t * TOPK + 1];
  int p2 = slotPos[t * TOPK + 2];
  int p3 = slotPos[t * TOPK + 3];
  for (int i = tid; i < HH; i += 256) {
    float acc = x2[(size_t)t * HH + i];
    acc += slotOut[(size_t)p0 * HH + i];
    acc += slotOut[(size_t)p1 * HH + i];
    acc += slotOut[(size_t)p2 * HH + i];
    acc += slotOut[(size_t)p3 * HH + i];
    out[(size_t)t * HH + i] = acc;
  }
}

// ---------------- host launch ----------------
extern "C" void kernel_launch(void* const* d_in, const int* in_sizes, int n_in,
                              void* d_out, int out_size, void* d_ws, size_t ws_size,
                              hipStream_t stream) {
  (void)in_sizes; (void)n_in; (void)out_size; (void)ws_size;
  const float* x     = (const float*)d_in[0];
  // d_in[1] attention_mask: causal, applied analytically in k_attn
  const float* ln1w  = (const float*)d_in[2];
  const float* ln2w  = (const float*)d_in[3];
  const float* qw    = (const float*)d_in[4];
  const float* qb    = (const float*)d_in[5];
  const float* kw    = (const float*)d_in[6];
  const float* kbias = (const float*)d_in[7];
  const float* vw    = (const float*)d_in[8];
  const float* vb    = (const float*)d_in[9];
  const float* ow    = (const float*)d_in[10];
  const float* obias = (const float*)d_in[11];
  const float* rw    = (const float*)d_in[12];
  const float* rb    = (const float*)d_in[13];
  const float* gupw  = (const float*)d_in[14];
  const float* gupb  = (const float*)d_in[15];
  const float* dw    = (const float*)d_in[16];
  const float* db    = (const float*)d_in[17];
  float* out = (float*)d_out;

  char* ws = (char*)d_ws;
  size_t off = 0;
  auto alloc = [&](size_t bytes) -> void* {
    void* p = ws + off;
    off = (off + bytes + 255) & ~(size_t)255;
    return p;
  };
  _Float16* h16      = (_Float16*)alloc((size_t)TT * HH * 2);
  float*    qkv      = (float*)   alloc((size_t)TT * QKVW * 4);
  _Float16* attn16   = (_Float16*)alloc((size_t)TT * NHEADS * HD * 2);
  float*    x2       = (float*)   alloc((size_t)TT * HH * 4);
  _Float16* h2       = (_Float16*)alloc((size_t)TT * HH * 2);
  int*      topidx   = (int*)     alloc((size_t)TT * TOPK * 4);
  float*    topw     = (float*)   alloc((size_t)TT * TOPK * 4);
  int*      counts   = (int*)     alloc(EE * 4);
  int*      cursor   = (int*)     alloc(EE * 4);
  int*      tileExp  = (int*)     alloc(MAXROWT * 4);
  int*      permTok  = (int*)     alloc(MAXPAD * 4);
  int*      slotPos  = (int*)     alloc(MAXSLOTS * 4);
  _Float16* inter16  = (_Float16*)alloc((size_t)MAXPAD * II * 2);
  float*    slotOut  = (float*)   alloc((size_t)MAXPAD * HH * 4);

  k_init   <<<1, 256, 0, stream>>>(counts, tileExp, permTok);
  k_rms1   <<<TT, 256, 0, stream>>>(x, ln1w, h16);
  k_qkv    <<<dim3(TT / 64, QKVW / 16), 32, 0, stream>>>(h16, qw, qb, kw, kbias, vw, vb, qkv);
  k_attn   <<<dim3(SS / 16, NHEADS, BB), 32, 0, stream>>>(qkv, attn16);
  k_oproj  <<<dim3(TT / 64, HH / 16), 32, 0, stream>>>(attn16, ow, obias, x, x2);
  k_router <<<TT, 256, 0, stream>>>(x2, ln2w, rw, rb, h2, topidx, topw, counts);
  k_offsets<<<1, 1, 0, stream>>>(counts, cursor, tileExp);
  k_scatter<<<MAXSLOTS / 256, 256, 0, stream>>>(topidx, cursor, permTok, slotPos);
  k_moe1   <<<dim3(MAXROWT, II / 16), 32, 0, stream>>>(h2, permTok, tileExp, gupw, gupb, inter16);
  k_moe2   <<<dim3(MAXROWT, HH / 16), 32, 0, stream>>>(inter16, permTok, tileExp, topw, dw, db, slotOut);
  k_combine<<<TT, 256, 0, stream>>>(x2, slotOut, slotPos, out);
}
